// StyleResNetBlock3d_34935263986184
// MI455X (gfx1250) — matrix operand order repaired and checked by reading, not compiled, not run on hardware
//
#include <hip/hip_runtime.h>

typedef __attribute__((ext_vector_type(16))) _Float16 v16h;
typedef __attribute__((ext_vector_type(8)))  _Float16 v8h;
typedef __attribute__((ext_vector_type(4)))  _Float16 v4h;
typedef __attribute__((ext_vector_type(2)))  _Float16 v2h;
typedef __attribute__((ext_vector_type(8)))  float    v8f;
typedef __attribute__((ext_vector_type(4)))  int      v4i;

#define NEG_SLOPE 0.01f

#if defined(__gfx1250__) && __has_builtin(__builtin_amdgcn_global_load_async_to_lds_b128) && \
    __has_builtin(__builtin_amdgcn_s_wait_asynccnt)
#define USE_ASYNC_LDS 1
#else
#define USE_ASYNC_LDS 0
#endif

__device__ __forceinline__ float lrelu(float v) { return v > 0.f ? v : NEG_SLOPE * v; }

#if USE_ASYNC_LDS
// Async DMA 16B global -> LDS (GLOBAL_LOAD_ASYNC_TO_LDS_B128, ASYNCcnt-tracked).
// Builtin signature (from toolchain diagnostic): (v4i AS1*, v4i AS3*, imm offset, imm cpol).
__device__ __forceinline__ void async_copy16(const _Float16* g, _Float16* l) {
  __builtin_amdgcn_global_load_async_to_lds_b128(
      (__attribute__((address_space(1))) v4i*)g,
      (__attribute__((address_space(3))) v4i*)l, 0, 0);
}
#endif

// A-matrix fragment (16x32 f16, M x K), row-major weights in LDS, lda halves.
// Lane<16: halves = K[0..15] (kb=0); lane>=16: K[16..31] (kb=8).
__device__ __forceinline__ v16h ld_a_frag(const _Float16* row, int kb) {
  union { v16h v; v8h h[2]; } u;
  u.h[0] = *(const v8h*)(row + 2 * kb);
  u.h[1] = *(const v8h*)(row + 2 * kb + 8);
  return u.v;
}
// B-matrix fragment (32x16 f16, K x N) from activation slab stored [n][k]:
// lane group g = lane>>4 supplies K = 16*g .. 16*g+15, contiguous in LDS.
__device__ __forceinline__ v16h ld_b_frag(const _Float16* p) {
  union { v16h v; v8h h[2]; } u;
  u.h[0] = *(const v8h*)(p);
  u.h[1] = *(const v8h*)(p + 8);
  return u.v;
}
__device__ __forceinline__ v8f wmma_f16(v16h a, v16h b, v8f c) {
  return __builtin_amdgcn_wmma_f32_16x16x32_f16(false, a, false, b, (short)0, c, false, false);
}

// ---------------------------------------------------------------------------
// Style MLPs: mods[0..63] = skip (2x32), [64..127] = c1 (2x32), [128..255] = c2 (2x64)
// ---------------------------------------------------------------------------
__device__ void mlp_eval(const float* sv, int din, int dh, int dout,
                         const float* w1, const float* b1,
                         const float* w2, const float* b2,
                         const float* w3, const float* b3,
                         float* out, float* sh1, float* sh2) {
  const int t = threadIdx.x;
  if (t < dh) {
    float a = b1[t];
    for (int k = 0; k < din; ++k) a += sv[k] * w1[t * din + k];
    sh1[t] = lrelu(a);
  }
  __syncthreads();
  if (t < dh) {
    float a = b2[t];
    for (int k = 0; k < dh; ++k) a += sh1[k] * w2[t * dh + k];
    sh2[t] = lrelu(a);
  }
  __syncthreads();
  if (t < dout) {
    float a = b3[t];
    for (int k = 0; k < dh; ++k) a += sh2[k] * w3[t * dh + k];
    out[t] = a;
  }
  __syncthreads();
}

__global__ void mods_kernel(const float* __restrict__ s,
    const float* sw1, const float* sb1, const float* sw2, const float* sb2,
    const float* sw3, const float* sb3,
    const float* aw1, const float* ab1, const float* aw2, const float* ab2,
    const float* aw3, const float* ab3,
    const float* bw1, const float* bb1, const float* bw2, const float* bb2,
    const float* bw3, const float* bb3,
    float* __restrict__ mods) {
  __shared__ float sv[64], sh1[128], sh2[128];
  for (int b = 0; b < 2; ++b) {
    if (threadIdx.x < 64) sv[threadIdx.x] = s[b * 64 + threadIdx.x];
    __syncthreads();
    mlp_eval(sv, 64, 64, 32,  sw1, sb1, sw2, sb2, sw3, sb3, mods + b * 32,        sh1, sh2);
    mlp_eval(sv, 64, 64, 32,  aw1, ab1, aw2, ab2, aw3, ab3, mods + 64 + b * 32,   sh1, sh2);
    mlp_eval(sv, 64, 128, 64, bw1, bb1, bw2, bb2, bw3, bb3, mods + 128 + b * 64,  sh1, sh2);
    __syncthreads();
  }
}

// ---------------------------------------------------------------------------
// Modulate + demodulate weights -> f16, layout [b][tap][o][cin]
// blockIdx.x = set*128 + b*64 + o ; set 0=skip(K=32), 1=c1(K=864), 2=c2(K=1728)
// ---------------------------------------------------------------------------
__global__ __launch_bounds__(256) void modw_kernel(
    const float* __restrict__ skip_wc, const float* __restrict__ c1_wc,
    const float* __restrict__ c2_wc, const float* __restrict__ mods,
    _Float16* __restrict__ wskip_h, _Float16* __restrict__ wc1_h,
    _Float16* __restrict__ wc2_h) {
  const int bid = blockIdx.x;
  const int o = bid & 63;
  const int b = (bid >> 6) & 1;
  const int set = bid >> 7;
  const int t = threadIdx.x;
  __shared__ float red[256];

  float ss = 0.f;
  if (set == 0) {
    if (t < 32) { float w = skip_wc[o * 32 + t] * mods[b * 32 + t]; ss = w * w; }
  } else if (set == 1) {
    for (int e = t; e < 864; e += 256) {
      int c = e / 27, tap = e % 27;
      float w = c1_wc[(o * 32 + c) * 27 + tap] * mods[64 + b * 32 + c];
      ss += w * w;
    }
  } else {
    for (int e = t; e < 1728; e += 256) {
      int c = e / 27, tap = e % 27;
      float w = c2_wc[(o * 64 + c) * 27 + tap] * mods[128 + b * 64 + c];
      ss += w * w;
    }
  }
  red[t] = ss;
  __syncthreads();
  for (int k = 128; k > 0; k >>= 1) {
    if (t < k) red[t] += red[t + k];
    __syncthreads();
  }
  const float scale = rsqrtf(red[0] + 1e-8f);

  if (set == 0) {
    if (t < 32)
      wskip_h[(size_t)(b * 64 + o) * 32 + t] =
          (_Float16)(skip_wc[o * 32 + t] * mods[b * 32 + t] * scale);
  } else if (set == 1) {
    for (int e = t; e < 864; e += 256) {
      int c = e / 27, tap = e % 27;
      wc1_h[((size_t)(b * 27 + tap) * 64 + o) * 32 + c] =
          (_Float16)(c1_wc[(o * 32 + c) * 27 + tap] * mods[64 + b * 32 + c] * scale);
    }
  } else {
    for (int e = t; e < 1728; e += 256) {
      int c = e / 27, tap = e % 27;
      wc2_h[((size_t)(b * 27 + tap) * 64 + o) * 64 + c] =
          (_Float16)(c2_wc[(o * 64 + c) * 27 + tap] * mods[128 + b * 64 + c] * scale);
    }
  }
}

// ---------------------------------------------------------------------------
// x f32 -> f16
// ---------------------------------------------------------------------------
__global__ void cvt_kernel(const float* __restrict__ x, _Float16* __restrict__ xh, size_t n4) {
  size_t i = (size_t)blockIdx.x * blockDim.x + threadIdx.x;
  const size_t stride = (size_t)gridDim.x * blockDim.x;
  for (; i < n4; i += stride) {
    float4 v = ((const float4*)x)[i];
    v4h o;
    o[0] = (_Float16)v.x; o[1] = (_Float16)v.y; o[2] = (_Float16)v.z; o[3] = (_Float16)v.w;
    ((v4h*)xh)[i] = o;
  }
}

// ---------------------------------------------------------------------------
// c1: 3x3x3 conv 32->64, x(96^3 f16) -> h(94^3 f16), fused bias + leaky relu
// Block: 64 outch x 64 W-points; 8 waves, each owns two 16x16 tiles.
// ---------------------------------------------------------------------------
__global__ __launch_bounds__(256) void c1_kernel(const _Float16* __restrict__ xh,
                                                 const _Float16* __restrict__ wtap,  // [b][tap][o][c32]
                                                 const float* __restrict__ bias,
                                                 _Float16* __restrict__ hh) {
  __shared__ __align__(16) _Float16 Wt[3 * 64 * 32];  // 3 kw taps, [o][c]
  __shared__ __align__(16) _Float16 Bt[68 * 32];      // [w_local][c]
  const int tid = threadIdx.x;
  const int lane = tid & 31;
  const int wave = tid >> 5;
  const int mt0 = (wave & 1) * 2;   // M tiles mt0, mt0+1
  const int nt = wave >> 1;         // N tile
  const int w0 = blockIdx.x * 64;
  const int hy = blockIdx.y;
  const int b = blockIdx.z / 94;
  const int dz = blockIdx.z % 94;

  const size_t xplane = 96 * 96;
  const size_t xchan = (size_t)96 * 96 * 96;
  const size_t xmax = (size_t)2 * 32 * xchan - 1;

  const int m = lane & 15;
  const int kb = (lane < 16) ? 0 : 8;
  const int grp = lane >> 4;

  v8f acc0 = {}; v8f acc1 = {};

  for (int kd = 0; kd < 3; ++kd) {
    for (int kh = 0; kh < 3; ++kh) {
      __syncthreads();
      {  // stage 3 kw weight taps (contiguous in source layout)
        const _Float16* src = wtap + (size_t)(b * 27 + (kd * 3 + kh) * 3) * (64 * 32);
#if USE_ASYNC_LDS
        for (int i = tid; i < (3 * 64 * 32) / 8; i += 256)
          async_copy16(src + (size_t)i * 8, Wt + (size_t)i * 8);
#else
        for (int i = tid; i < (3 * 64 * 32) / 8; i += 256)
          ((v8h*)Wt)[i] = ((const v8h*)src)[i];
#endif
      }
      {  // stage activation slab [w_local 0..67][c 0..31], 2-half vector chunks
        const int di = dz + kd, hi = hy + kh;
        for (int i = tid; i < 32 * 34; i += 256) {
          int c = i / 34, wl = (i % 34) * 2;
          size_t g = (size_t)(b * 32 + c) * xchan + (size_t)di * xplane +
                     (size_t)hi * 96 + (size_t)(w0 + wl);
          if (g > xmax - 1) g = xmax - 1;  // masked at store time
          v2h v = *(const v2h*)(xh + g);
          Bt[(wl + 0) * 32 + c] = v[0];
          Bt[(wl + 1) * 32 + c] = v[1];
        }
      }
#if USE_ASYNC_LDS
      __builtin_amdgcn_s_wait_asynccnt(0);
#endif
      __syncthreads();
      for (int kw = 0; kw < 3; ++kw) {
        const _Float16* Wk = Wt + kw * (64 * 32);
        v16h a0 = ld_a_frag(Wk + (mt0 * 16 + m) * 32, kb);
        v16h a1 = ld_a_frag(Wk + ((mt0 + 1) * 16 + m) * 32, kb);
        v16h bf = ld_b_frag(Bt + (nt * 16 + m + kw) * 32 + grp * 16);
        acc0 = wmma_f16(a0, bf, acc0);
        acc1 = wmma_f16(a1, bf, acc1);
      }
    }
  }

  const int wcol = w0 + nt * 16 + m;
  if (wcol < 94) {
    const int m_lo = grp * 8;
    for (int r = 0; r < 8; ++r) {
      int o0 = mt0 * 16 + m_lo + r;
      int o1 = o0 + 16;
      size_t i0 = (((size_t)(b * 64 + o0) * 94 + dz) * 94 + hy) * 94 + wcol;
      size_t i1 = (((size_t)(b * 64 + o1) * 94 + dz) * 94 + hy) * 94 + wcol;
      hh[i0] = (_Float16)lrelu(acc0[r] + bias[o0]);
      hh[i1] = (_Float16)lrelu(acc1[r] + bias[o1]);
    }
  }
}

// ---------------------------------------------------------------------------
// c2: 3x3x3 conv 64->64 on h(94^3) -> 92^3, fused skip(1x1 conv of x at +2 crop)
// + both biases + leaky relu -> f32 output.
// ---------------------------------------------------------------------------
__global__ __launch_bounds__(256) void c2_kernel(const _Float16* __restrict__ hh,
                                                 const _Float16* __restrict__ xh,
                                                 const _Float16* __restrict__ wtap,   // [b][tap][o][c64]
                                                 const _Float16* __restrict__ wskip,  // [b][o][c32]
                                                 const float* __restrict__ bias2,
                                                 const float* __restrict__ biass,
                                                 float* __restrict__ out) {
  __shared__ __align__(16) _Float16 Wt[3 * 64 * 64];  // 24 KB
  __shared__ __align__(16) _Float16 Bt[68 * 64];      // 8.5 KB
  __shared__ __align__(16) _Float16 Ws[64 * 32];      // skip weights
  __shared__ __align__(16) _Float16 Bs[64 * 32];      // skip activations [w][c]
  const int tid = threadIdx.x;
  const int lane = tid & 31;
  const int wave = tid >> 5;
  const int mt0 = (wave & 1) * 2;
  const int nt = wave >> 1;
  const int w0 = blockIdx.x * 64;
  const int hy = blockIdx.y;
  const int b = blockIdx.z / 92;
  const int dz = blockIdx.z % 92;

  const size_t hplane = 94 * 94;
  const size_t hchan = (size_t)94 * 94 * 94;
  const size_t hmax = (size_t)2 * 64 * hchan - 1;
  const size_t xplane = 96 * 96;
  const size_t xchan = (size_t)96 * 96 * 96;
  const size_t xmax = (size_t)2 * 32 * xchan - 1;

  const int m = lane & 15;
  const int kb = (lane < 16) ? 0 : 8;
  const int grp = lane >> 4;

  {  // stage skip weights + skip activations once (visible after first barrier)
    const _Float16* src = wskip + (size_t)b * (64 * 32);
#if USE_ASYNC_LDS
    for (int i = tid; i < (64 * 32) / 8; i += 256)
      async_copy16(src + (size_t)i * 8, Ws + (size_t)i * 8);
#else
    for (int i = tid; i < (64 * 32) / 8; i += 256)
      ((v8h*)Ws)[i] = ((const v8h*)src)[i];
#endif
    const int di = dz + 2, hi = hy + 2;
    for (int i = tid; i < 32 * 32; i += 256) {
      int c = i >> 5, wl = (i & 31) * 2;
      size_t g = (size_t)(b * 32 + c) * xchan + (size_t)di * xplane +
                 (size_t)hi * 96 + (size_t)(w0 + 2 + wl);
      if (g > xmax - 1) g = xmax - 1;
      v2h v = *(const v2h*)(xh + g);
      Bs[(wl + 0) * 32 + c] = v[0];
      Bs[(wl + 1) * 32 + c] = v[1];
    }
  }

  v8f acc0 = {}; v8f acc1 = {};

  for (int kd = 0; kd < 3; ++kd) {
    for (int kh = 0; kh < 3; ++kh) {
      __syncthreads();
      {  // stage 3 kw weight taps (64x64 each)
        const _Float16* src = wtap + (size_t)(b * 27 + (kd * 3 + kh) * 3) * (64 * 64);
#if USE_ASYNC_LDS
        for (int i = tid; i < (3 * 64 * 64) / 8; i += 256)
          async_copy16(src + (size_t)i * 8, Wt + (size_t)i * 8);
#else
        for (int i = tid; i < (3 * 64 * 64) / 8; i += 256)
          ((v8h*)Wt)[i] = ((const v8h*)src)[i];
#endif
      }
      {  // stage activation slab [w_local 0..67][c 0..63], 2-half vector chunks
        const int di = dz + kd, hi = hy + kh;
        for (int i = tid; i < 64 * 34; i += 256) {
          int c = i / 34, wl = (i % 34) * 2;
          size_t g = (size_t)(b * 64 + c) * hchan + (size_t)di * hplane +
                     (size_t)hi * 94 + (size_t)(w0 + wl);
          if (g > hmax - 1) g = hmax - 1;
          v2h v = *(const v2h*)(hh + g);
          Bt[(wl + 0) * 64 + c] = v[0];
          Bt[(wl + 1) * 64 + c] = v[1];
        }
      }
#if USE_ASYNC_LDS
      __builtin_amdgcn_s_wait_asynccnt(0);
#endif
      __syncthreads();
      for (int kw = 0; kw < 3; ++kw) {
        const _Float16* Wk = Wt + kw * (64 * 64);
        for (int ks = 0; ks < 2; ++ks) {  // K = 64 -> two K=32 WMMA steps
          v16h a0 = ld_a_frag(Wk + (mt0 * 16 + m) * 64 + ks * 32, kb);
          v16h a1 = ld_a_frag(Wk + ((mt0 + 1) * 16 + m) * 64 + ks * 32, kb);
          v16h bf = ld_b_frag(Bt + (nt * 16 + m + kw) * 64 + ks * 32 + grp * 16);
          acc0 = wmma_f16(a0, bf, acc0);
          acc1 = wmma_f16(a1, bf, acc1);
        }
      }
    }
  }

  {  // fused skip path: one K=32 WMMA per tile
    v16h a0 = ld_a_frag(Ws + (mt0 * 16 + m) * 32, kb);
    v16h a1 = ld_a_frag(Ws + ((mt0 + 1) * 16 + m) * 32, kb);
    v16h bf = ld_b_frag(Bs + (nt * 16 + m) * 32 + grp * 16);
    acc0 = wmma_f16(a0, bf, acc0);
    acc1 = wmma_f16(a1, bf, acc1);
  }

  const int wcol = w0 + nt * 16 + m;
  if (wcol < 92) {
    const int m_lo = grp * 8;
    for (int r = 0; r < 8; ++r) {
      int o0 = mt0 * 16 + m_lo + r;
      int o1 = o0 + 16;
      size_t i0 = (((size_t)(b * 64 + o0) * 92 + dz) * 92 + hy) * 92 + wcol;
      size_t i1 = (((size_t)(b * 64 + o1) * 92 + dz) * 92 + hy) * 92 + wcol;
      out[i0] = lrelu(acc0[r] + bias2[o0] + biass[o0]);
      out[i1] = lrelu(acc1[r] + bias2[o1] + biass[o1]);
    }
  }
}

// ---------------------------------------------------------------------------
extern "C" void kernel_launch(void* const* d_in, const int* in_sizes, int n_in,
                              void* d_out, int out_size, void* d_ws, size_t ws_size,
                              hipStream_t stream) {
  (void)in_sizes; (void)n_in; (void)out_size; (void)ws_size;
  const float* x = (const float*)d_in[0];
  const float* s = (const float*)d_in[1];

  char* ws = (char*)d_ws;
  size_t off = 0;
  auto take = [&](size_t bytes) -> char* {
    char* p = ws + off;
    off = (off + bytes + 255) & ~(size_t)255;
    return p;
  };
  float*    mods    = (float*)take(256 * sizeof(float));
  _Float16* wskip_h = (_Float16*)take((size_t)2 * 64 * 32 * 2);
  _Float16* wc1_h   = (_Float16*)take((size_t)2 * 27 * 64 * 32 * 2);
  _Float16* wc2_h   = (_Float16*)take((size_t)2 * 27 * 64 * 64 * 2);
  _Float16* xh      = (_Float16*)take((size_t)56623104 * 2);   // 2*32*96^3
  _Float16* hh      = (_Float16*)take((size_t)106314752 * 2);  // 2*64*94^3

  mods_kernel<<<1, 128, 0, stream>>>(
      s,
      (const float*)d_in[2],  (const float*)d_in[3],  (const float*)d_in[4],
      (const float*)d_in[5],  (const float*)d_in[6],  (const float*)d_in[7],
      (const float*)d_in[10], (const float*)d_in[11], (const float*)d_in[12],
      (const float*)d_in[13], (const float*)d_in[14], (const float*)d_in[15],
      (const float*)d_in[18], (const float*)d_in[19], (const float*)d_in[20],
      (const float*)d_in[21], (const float*)d_in[22], (const float*)d_in[23],
      mods);

  modw_kernel<<<384, 256, 0, stream>>>(
      (const float*)d_in[8], (const float*)d_in[16], (const float*)d_in[24],
      mods, wskip_h, wc1_h, wc2_h);

  cvt_kernel<<<2048, 256, 0, stream>>>(x, xh, (size_t)56623104 / 4);

  c1_kernel<<<dim3(2, 94, 2 * 94), 256, 0, stream>>>(
      xh, wc1_h, (const float*)d_in[17], hh);

  c2_kernel<<<dim3(2, 92, 2 * 92), 256, 0, stream>>>(
      hh, xh, wc2_h, wskip_h, (const float*)d_in[25], (const float*)d_in[9],
      (float*)d_out);
}
